// AttentionInteraction_21303037788809
// MI455X (gfx1250) — compile-verified, matrix operand order checked by reading, not run on hardware
//
#include <hip/hip_runtime.h>
#include <hip/hip_bf16.h>
#include <math.h>

// ---------------------------------------------------------------------------
// CDNA5 / gfx1250, wave32.  D = A*B + C via V_WMMA_F32_16X16X32_BF16.
// Fully fused per-graph attention (attention is block-diagonal per graph, so
// q/k/v never round-trip through HBM; only h is read and outputs written).
// ---------------------------------------------------------------------------

typedef __attribute__((ext_vector_type(16))) __bf16 v16bf;
typedef __attribute__((ext_vector_type(8)))  float  v8f;

#define GRAPHS 512
#define NADS   16
#define NCAT   256
#define DIM    256
#define SCALE  0.0625f   // 1/sqrt(256)

// A/B 16-bit fragment K mapping (ISA 7.12.2):
//   half h (0..15): g = h/8, j = h%8, K = kbase + 16*g + khi + j,
//   khi = 8 for lanes 16..31 else 0.  Row (A: M, B: N) = lane%16; caller
// passes a pointer to the start of that row.
__device__ __forceinline__ v16bf frag_f32(const float* row, int kbase, int khi) {
  v16bf f;
  const float* p = row + kbase + khi;
#pragma unroll
  for (int j = 0; j < 8; ++j) f[j] = (__bf16)p[j];
#pragma unroll
  for (int j = 0; j < 8; ++j) f[8 + j] = (__bf16)p[16 + j];
  return f;
}

__device__ __forceinline__ v16bf frag_bf16(const __bf16* row, int kbase, int khi) {
  v16bf f;
  const __bf16* p = row + kbase + khi;
#pragma unroll
  for (int j = 0; j < 8; ++j) f[j] = p[j];
#pragma unroll
  for (int j = 0; j < 8; ++j) f[8 + j] = p[16 + j];
  return f;
}

__device__ __forceinline__ v8f wmma_bf16(v16bf a, v16bf b, v8f c) {
  return __builtin_amdgcn_wmma_f32_16x16x32_bf16(false, a, false, b,
                                                 (short)0, c, false, false);
}

// ---------------------------------------------------------------------------
// Weight f32 -> bf16 one-time conversion into workspace (6 x 256x256).
// ---------------------------------------------------------------------------
__global__ void cvt_weights_kernel(const float* w0, const float* w1,
                                   const float* w2, const float* w3,
                                   const float* w4, const float* w5,
                                   __bf16* dst) {
  int w = blockIdx.y;
  int i = blockIdx.x * blockDim.x + threadIdx.x;  // 0..65535
  const float* src;
  switch (w) {
    case 0:  src = w0; break;
    case 1:  src = w1; break;
    case 2:  src = w2; break;
    case 3:  src = w3; break;
    case 4:  src = w4; break;
    default: src = w5; break;
  }
  dst[(size_t)w * DIM * DIM + i] = (__bf16)src[i];
}

// ---------------------------------------------------------------------------
// Kernel A: new_h_ads for one graph per workgroup.
// ---------------------------------------------------------------------------
__global__ __launch_bounds__(256, 1)
void ads_update_kernel(const float* __restrict__ hAall,
                       const float* __restrict__ hCall,
                       const __bf16* __restrict__ Wq, const float* __restrict__ bq,
                       const __bf16* __restrict__ Wk, const float* __restrict__ bk,
                       const __bf16* __restrict__ Wv, const float* __restrict__ bv,
                       float* __restrict__ out) {
  __shared__ __bf16 qa[NADS][DIM];   // 8KB: q, later att (bf16)
  __shared__ float  sc[NADS][DIM];   // 16KB: scores, later residual+upd accum
  __shared__ __bf16 wbuf[8][8192];   // 128KB: per-wave k-tile [16][256] / vT [256][32]
  __shared__ float  red[NADS][16];
  __shared__ float  stat[NADS];

  const int g    = blockIdx.x;
  const int tid  = threadIdx.x;
  const int lane = tid & 31;
  const int wave = tid >> 5;
  const int l16  = lane & 15;
  const int grp  = lane >> 4;          // 0 / 1
  const int khi  = grp ? 8 : 0;
  const float* hA = hAall + (size_t)g * NADS * DIM;
  const float* hC = hCall + (size_t)g * NCAT * DIM;
  __bf16* kt = &wbuf[wave][0];

  // ---- Phase 1: q = hA @ Wq^T + bq  (2 n-tiles per wave) ----
  {
    v16bf afr[8];
#pragma unroll
    for (int kc = 0; kc < 8; ++kc)
      afr[kc] = frag_f32(hA + l16 * DIM, kc * 32, khi);   // load/convert ONCE
#pragma unroll
    for (int t = 0; t < 2; ++t) {
      int nt = wave * 2 + t;
      v8f acc = {};
#pragma unroll
      for (int kc = 0; kc < 8; ++kc) {
        v16bf b = frag_bf16(Wq + (size_t)(nt * 16 + l16) * DIM, kc * 32, khi);
        acc = wmma_bf16(afr[kc], b, acc);
      }
      int dcol = nt * 16 + l16;
      float bias = bq[dcol];
#pragma unroll
      for (int r = 0; r < 8; ++r)
        qa[r + grp * 8][dcol] = (__bf16)(acc[r] + bias);
    }
  }
  __syncthreads();

  // ---- Phase 2: per-wave c-tiles: k tile then scores ----
#pragma unroll 1
  for (int t = 0; t < 2; ++t) {
    int ct = wave * 2 + t;
    __builtin_prefetch(hC + (size_t)(ct * 16 + l16) * DIM, 0, 1);
    {
      v16bf afr[8];
#pragma unroll
      for (int kc = 0; kc < 8; ++kc)
        afr[kc] = frag_f32(hC + (size_t)(ct * 16 + l16) * DIM, kc * 32, khi);
      for (int nt = 0; nt < 16; ++nt) {
        v8f acc = {};
#pragma unroll
        for (int kc = 0; kc < 8; ++kc) {
          v16bf b = frag_bf16(Wk + (size_t)(nt * 16 + l16) * DIM, kc * 32, khi);
          acc = wmma_bf16(afr[kc], b, acc);
        }
        int dcol = nt * 16 + l16;
        float bias = bk[dcol];
#pragma unroll
        for (int r = 0; r < 8; ++r)
          kt[(r + grp * 8) * DIM + dcol] = (__bf16)(acc[r] + bias);
      }
    }
    __syncthreads();
    // scores tile [16 a][16 c]
    v8f acc = {};
#pragma unroll
    for (int kc = 0; kc < 8; ++kc) {
      v16bf a = frag_bf16(&qa[l16][0], kc * 32, khi);
      v16bf b = frag_bf16(&kt[l16 * DIM], kc * 32, khi);
      acc = wmma_bf16(a, b, acc);
    }
#pragma unroll
    for (int r = 0; r < 8; ++r)
      sc[r + grp * 8][ct * 16 + l16] = acc[r] * SCALE;
    __syncthreads();
  }

  // ---- Softmax over 256 keys per ads row; att (bf16) into qa ----
  {
    int row = tid & 15, seg = tid >> 4;
    float m = -3.4e38f;
#pragma unroll
    for (int j = 0; j < 16; ++j) m = fmaxf(m, sc[row][seg * 16 + j]);
    red[row][seg] = m;
    __syncthreads();
    if (tid < 16) {
      float mm = red[tid][0];
      for (int s2 = 1; s2 < 16; ++s2) mm = fmaxf(mm, red[tid][s2]);
      stat[tid] = mm;
    }
    __syncthreads();
    float mx = stat[row];
    float ssum = 0.f;
#pragma unroll
    for (int j = 0; j < 16; ++j) {
      float e = __expf(sc[row][seg * 16 + j] - mx);
      sc[row][seg * 16 + j] = e;
      ssum += e;
    }
    red[row][seg] = ssum;
    __syncthreads();
    if (tid < 16) {
      float s2v = 0.f;
      for (int s2 = 0; s2 < 16; ++s2) s2v += red[tid][s2];
      stat[tid] = 1.0f / s2v;
    }
    __syncthreads();
    float inv = stat[row];
#pragma unroll
    for (int j = 0; j < 16; ++j)
      qa[row][seg * 16 + j] = (__bf16)(sc[row][seg * 16 + j] * inv);
  }
  __syncthreads();

  // ---- Re-init sc with residual base hA ----
  {
    int row = tid & 15, seg = tid >> 4;
#pragma unroll
    for (int j = 0; j < 16; ++j)
      sc[row][seg * 16 + j] = hA[row * DIM + seg * 16 + j];
  }
  __syncthreads();

  // ---- Phase 3: v tiles for this wave's 32 c values, then partial upd ----
  __bf16* vt = &wbuf[wave][0];  // [256 d][32 c-local]
#pragma unroll 1
  for (int t = 0; t < 2; ++t) {
    int ct = wave * 2 + t;
    v16bf afr[8];
#pragma unroll
    for (int kc = 0; kc < 8; ++kc)
      afr[kc] = frag_f32(hC + (size_t)(ct * 16 + l16) * DIM, kc * 32, khi);
    for (int nt = 0; nt < 16; ++nt) {
      v8f acc = {};
#pragma unroll
      for (int kc = 0; kc < 8; ++kc) {
        v16bf b = frag_bf16(Wv + (size_t)(nt * 16 + l16) * DIM, kc * 32, khi);
        acc = wmma_bf16(afr[kc], b, acc);
      }
      int dcol = nt * 16 + l16;
      float bias = bv[dcol];
#pragma unroll
      for (int r = 0; r < 8; ++r)
        vt[dcol * 32 + t * 16 + r + grp * 8] = (__bf16)(acc[r] + bias);
    }
  }
  __syncthreads();

  // upd partial: this wave's K-chunk of 32 cat nodes (c = wave*32 .. +31)
  {
    v16bf a = frag_bf16(&qa[l16][wave * 32], 0, khi);
    for (int dt = 0; dt < 16; ++dt) {
      v16bf b = frag_bf16(&vt[(dt * 16 + l16) * 32], 0, khi);
      v8f u = {};
      u = wmma_bf16(a, b, u);
      int dcol = dt * 16 + l16;
#pragma unroll
      for (int r = 0; r < 8; ++r)
        atomicAdd(&sc[r + grp * 8][dcol], u[r]);   // ds_add_f32
    }
  }
  __syncthreads();

  // ---- L2 normalize rows of (hA + upd), write out ----
  {
    int row = tid & 15, seg = tid >> 4;
    float ssq = 0.f;
#pragma unroll
    for (int j = 0; j < 16; ++j) {
      float x = sc[row][seg * 16 + j];
      ssq += x * x;
    }
    red[row][seg] = ssq;
    __syncthreads();
    if (tid < 16) {
      float s2v = 0.f;
      for (int s2 = 0; s2 < 16; ++s2) s2v += red[tid][s2];
      float n = sqrtf(s2v);
      stat[tid] = 1.0f / fmaxf(n, 1e-12f);
    }
    __syncthreads();
    float inv = stat[row];
    float* og = out + (size_t)g * NADS * DIM;
#pragma unroll
    for (int j = 0; j < 16; ++j)
      og[row * DIM + seg * 16 + j] = sc[row][seg * 16 + j] * inv;
  }
}

// ---------------------------------------------------------------------------
// Kernel B: new_h_cat for one graph per workgroup.
// ---------------------------------------------------------------------------
__global__ __launch_bounds__(256, 1)
void cat_update_kernel(const float* __restrict__ hAall,
                       const float* __restrict__ hCall,
                       const __bf16* __restrict__ Wq, const float* __restrict__ bq,
                       const __bf16* __restrict__ Wk, const float* __restrict__ bk,
                       const __bf16* __restrict__ Wv, const float* __restrict__ bv,
                       float* __restrict__ out) {
  __shared__ __bf16 kA[NADS][DIM];     // 8KB  k_ads row-major [a][d]
  __shared__ __bf16 vAT[DIM][32];      // 16KB v_ads^T [d][a] padded to 32
  __shared__ __bf16 qbuf[8][16][DIM];  // 64KB per-wave q_cat tile
  __shared__ __bf16 attb[8][16][32];   // 8KB  per-wave att, K padded to 32

  const int g    = blockIdx.x;
  const int tid  = threadIdx.x;
  const int lane = tid & 31;
  const int wave = tid >> 5;
  const int l16  = lane & 15;
  const int grp  = lane >> 4;
  const int khi  = grp ? 8 : 0;
  const float* hA = hAall + (size_t)g * NADS * DIM;
  const float* hC = hCall + (size_t)g * NCAT * DIM;

  // ---- Phase 1: kA and vA (32 tile jobs, 4 per wave) ----
  {
    v16bf afr[8];
#pragma unroll
    for (int kc = 0; kc < 8; ++kc)
      afr[kc] = frag_f32(hA + l16 * DIM, kc * 32, khi);   // shared by all jobs
#pragma unroll 1
    for (int t = 0; t < 4; ++t) {
      int job = wave * 4 + t;
      int nt = job & 15;
      bool isv = (job >= 16);
      const __bf16* W = isv ? Wv : Wk;
      const float*  bb = isv ? bv : bk;
      v8f acc = {};
#pragma unroll
      for (int kc = 0; kc < 8; ++kc) {
        v16bf b = frag_bf16(W + (size_t)(nt * 16 + l16) * DIM, kc * 32, khi);
        acc = wmma_bf16(afr[kc], b, acc);
      }
      int dcol = nt * 16 + l16;
      float bias = bb[dcol];
      if (!isv) {
#pragma unroll
        for (int r = 0; r < 8; ++r)
          kA[r + grp * 8][dcol] = (__bf16)(acc[r] + bias);
      } else {
#pragma unroll
        for (int r = 0; r < 8; ++r)
          vAT[dcol][r + grp * 8] = (__bf16)(acc[r] + bias);
      }
    }
  }
  // zero-pad vAT columns a=16..31 (K padding for the K=16 attention GEMM)
  {
    int d = tid;
#pragma unroll
    for (int a2 = 16; a2 < 32; ++a2) vAT[d][a2] = (__bf16)0.0f;
  }
  __syncthreads();

  // ---- Phase 2: per-wave c-tiles ----
#pragma unroll 1
  for (int t = 0; t < 2; ++t) {
    int ct = wave * 2 + t;
    // q_cat tile (16 rows x 256)
    {
      v16bf afr[8];
#pragma unroll
      for (int kc = 0; kc < 8; ++kc)
        afr[kc] = frag_f32(hC + (size_t)(ct * 16 + l16) * DIM, kc * 32, khi);
      for (int nt = 0; nt < 16; ++nt) {
        v8f acc = {};
#pragma unroll
        for (int kc = 0; kc < 8; ++kc) {
          v16bf b = frag_bf16(Wq + (size_t)(nt * 16 + l16) * DIM, kc * 32, khi);
          acc = wmma_bf16(afr[kc], b, acc);
        }
        int dcol = nt * 16 + l16;
        float bias = bq[dcol];
#pragma unroll
        for (int r = 0; r < 8; ++r)
          qbuf[wave][r + grp * 8][dcol] = (__bf16)(acc[r] + bias);
      }
    }
    __syncthreads();

    // scores 16(c) x 16(a)
    v8f s = {};
#pragma unroll
    for (int kc = 0; kc < 8; ++kc) {
      v16bf a = frag_bf16(&qbuf[wave][l16][0], kc * 32, khi);
      v16bf b = frag_bf16(&kA[l16][0], kc * 32, khi);
      s = wmma_bf16(a, b, s);
    }
    // softmax over the 16 ads keys = across 16 lanes of each half-wave
#pragma unroll
    for (int r = 0; r < 8; ++r) {
      float x = s[r] * SCALE;
      float m = x;
      for (int off = 1; off < 16; off <<= 1) m = fmaxf(m, __shfl_xor(m, off, 16));
      float e = __expf(x - m);
      float sum = e;
      for (int off = 1; off < 16; off <<= 1) sum += __shfl_xor(sum, off, 16);
      attb[wave][r + grp * 8][l16] = (__bf16)(e / sum);
      attb[wave][r + grp * 8][16 + l16] = (__bf16)0.0f;  // K pad
    }
    __syncthreads();

    // upd pass 1: sum of squares of (hC + att*vA) per row (for L2 norm)
    float ssq[8];
#pragma unroll
    for (int r = 0; r < 8; ++r) ssq[r] = 0.f;
    {
      v16bf a = frag_bf16(&attb[wave][l16][0], 0, khi);
      for (int dt = 0; dt < 16; ++dt) {
        v16bf b = frag_bf16(&vAT[dt * 16 + l16][0], 0, khi);
        v8f u = {};
        u = wmma_bf16(a, b, u);
#pragma unroll
        for (int r = 0; r < 8; ++r) {
          int c_ = ct * 16 + r + grp * 8;
          int d_ = dt * 16 + l16;
          float x = u[r] + hC[(size_t)c_ * DIM + d_];
          ssq[r] += x * x;
        }
      }
    }
    float inv_[8];
#pragma unroll
    for (int r = 0; r < 8; ++r) {
      float v = ssq[r];
      for (int off = 1; off < 16; off <<= 1) v += __shfl_xor(v, off, 16);
      float n = sqrtf(v);
      inv_[r] = 1.0f / fmaxf(n, 1e-12f);
    }
    // upd pass 2: recompute (K=32, cheap), normalize, store to global
    {
      v16bf a = frag_bf16(&attb[wave][l16][0], 0, khi);
      for (int dt = 0; dt < 16; ++dt) {
        v16bf b = frag_bf16(&vAT[dt * 16 + l16][0], 0, khi);
        v8f u = {};
        u = wmma_bf16(a, b, u);
#pragma unroll
        for (int r = 0; r < 8; ++r) {
          int c_ = ct * 16 + r + grp * 8;
          int d_ = dt * 16 + l16;
          float x = u[r] + hC[(size_t)c_ * DIM + d_];
          out[((size_t)g * NCAT + c_) * DIM + d_] = x * inv_[r];
        }
      }
    }
    __syncthreads();
  }
}

// ---------------------------------------------------------------------------
// Launch
// ---------------------------------------------------------------------------
extern "C" void kernel_launch(void* const* d_in, const int* in_sizes, int n_in,
                              void* d_out, int out_size, void* d_ws, size_t ws_size,
                              hipStream_t stream) {
  (void)in_sizes; (void)n_in; (void)out_size; (void)ws_size;

  const float* h_ads = (const float*)d_in[0];
  const float* h_cat = (const float*)d_in[1];
  const float* WqA = (const float*)d_in[2];  const float* bqA = (const float*)d_in[3];
  const float* WkA = (const float*)d_in[4];  const float* bkA = (const float*)d_in[5];
  const float* WvA = (const float*)d_in[6];  const float* bvA = (const float*)d_in[7];
  const float* WqC = (const float*)d_in[8];  const float* bqC = (const float*)d_in[9];
  const float* WkC = (const float*)d_in[10]; const float* bkC = (const float*)d_in[11];
  const float* WvC = (const float*)d_in[12]; const float* bvC = (const float*)d_in[13];
  float* out = (float*)d_out;

  __bf16* W16 = (__bf16*)d_ws;                 // 6 x 65536 bf16 = 768 KB
  __bf16* WqA16 = W16 + 0 * 65536;
  __bf16* WkA16 = W16 + 1 * 65536;
  __bf16* WvA16 = W16 + 2 * 65536;
  __bf16* WqC16 = W16 + 3 * 65536;
  __bf16* WkC16 = W16 + 4 * 65536;
  __bf16* WvC16 = W16 + 5 * 65536;

  cvt_weights_kernel<<<dim3(256, 6), 256, 0, stream>>>(
      WqA, WkA, WvA, WqC, WkC, WvC, W16);

  ads_update_kernel<<<GRAPHS, 256, 0, stream>>>(
      h_ads, h_cat, WqA16, bqA, WkC16, bkC, WvC16, bvC, out);

  cat_update_kernel<<<GRAPHS, 256, 0, stream>>>(
      h_ads, h_cat, WqC16, bqC, WkA16, bkA, WvA16, bvA,
      out + (size_t)GRAPHS * NADS * DIM);
}
